// _STFTFn_45268955300551
// MI455X (gfx1250) — matrix operand order repaired and checked by reading, not compiled, not run on hardware
//
#include <hip/hip_runtime.h>

// ---------------- problem constants (from reference) ----------------
#define NBATCH   16
#define SAMPLES  262144
#define FL       1024          // filter length / win_length
#define HOP      256
#define LPAD     (FL - HOP)    // 768 zeros on the left
#define NFRAMES  1024          // frames per batch
#define NFREQ    513           // FL/2 + 1
#define NROWS    (2 * NFREQ)   // 1026 basis rows
#define MAGSZ    (NBATCH * NFREQ * NFRAMES)

#define KC       64            // K-chunk staged in LDS per iteration
#define MT       32            // freq rows per workgroup (real) (+32 imag)
#define NT       128           // frames per workgroup
#define NITER    (FL / KC)     // 16 K-chunks
#define AOPS     12            // async copies issued per thread per stage (uniform)

// element counts for the preconverted bf16 planes in d_ws
#define YN       (NBATCH * SAMPLES)   // 4,194,304
#define BN       (NROWS * FL)         // 1,050,624
#define WS_ELEMS (2 * (YN + BN))      // hi+lo planes for y and basis
// layout in d_ws (ushort elements): [0,YN) yhi | [YN,2YN) ylo | bhi | blo

typedef __attribute__((ext_vector_type(16))) __bf16 v16bf;
typedef __attribute__((ext_vector_type(8)))  float  v8f;
typedef __attribute__((ext_vector_type(4)))  int    v4i;

union Frag { uint4 u[2]; v16bf bf; };

#if __has_builtin(__builtin_amdgcn_global_load_async_to_lds_b128)
#define USE_ASYNC_LDS 1
// builtin signature (from hipcc diagnostic): (int4 AS1*, int4 AS3*, Imm, Imm)
typedef v4i __attribute__((address_space(1))) GV4;
typedef v4i __attribute__((address_space(3))) LV4;
#else
#define USE_ASYNC_LDS 0
#endif

// ---------------- helpers ----------------
static __device__ __forceinline__ unsigned short bf16_rn(float x) {
    unsigned u = __float_as_uint(x);
    return (unsigned short)((u + 0x7FFFu + ((u >> 16) & 1u)) >> 16);
}

static __device__ __forceinline__ void split4(float4 v, uint2& hp, uint2& lp) {
    unsigned short h0 = bf16_rn(v.x), h1 = bf16_rn(v.y);
    unsigned short h2 = bf16_rn(v.z), h3 = bf16_rn(v.w);
    float r0 = v.x - __uint_as_float((unsigned)h0 << 16);
    float r1 = v.y - __uint_as_float((unsigned)h1 << 16);
    float r2 = v.z - __uint_as_float((unsigned)h2 << 16);
    float r3 = v.w - __uint_as_float((unsigned)h3 << 16);
    unsigned short l0 = bf16_rn(r0), l1 = bf16_rn(r1);
    unsigned short l2 = bf16_rn(r2), l3 = bf16_rn(r3);
    hp = make_uint2((unsigned)h0 | ((unsigned)h1 << 16), (unsigned)h2 | ((unsigned)h3 << 16));
    lp = make_uint2((unsigned)l0 | ((unsigned)l1 << 16), (unsigned)l2 | ((unsigned)l3 << 16));
}

// copy 8 bf16 (16B) global -> LDS; async (no VGPR transit) when available
static __device__ __forceinline__ void copy16(unsigned short* lds, const unsigned short* g) {
#if USE_ASYNC_LDS
    __builtin_amdgcn_global_load_async_to_lds_b128(
        (GV4*)(void*)const_cast<unsigned short*>(g), (LV4*)(void*)lds, 0, 0);
#else
    *reinterpret_cast<uint4*>(lds) = *reinterpret_cast<const uint4*>(g);
#endif
}

static __device__ __forceinline__ void zero16(unsigned short* lds) {
    *reinterpret_cast<uint4*>(lds) = make_uint4(0u, 0u, 0u, 0u);
}

template <int N>
static __device__ __forceinline__ void wait_asyncN() {
#if USE_ASYNC_LDS
#if __has_builtin(__builtin_amdgcn_s_wait_asynccnt)
    __builtin_amdgcn_s_wait_asynccnt(N);
#else
    asm volatile("s_wait_asynccnt %0" :: "i"(N) : "memory");
#endif
#endif
}

// A fragment (16x32 bf16): lane L -> row M=L%16; lanes<16: K k0..k0+15 contiguous
static __device__ __forceinline__ v16bf ldfragA(const unsigned short* __restrict__ base,
                                                int row, int k0) {
    Frag f;
    f.u[0] = *reinterpret_cast<const uint4*>(&base[row * KC + k0]);
    f.u[1] = *reinterpret_cast<const uint4*>(&base[row * KC + k0 + 8]);
    return f.bf;
}
// B fragment (32x16 bf16): lane L -> col N=L%16; lanes<16: K k0..k0+15 contiguous
static __device__ __forceinline__ v16bf ldfragB(const unsigned short* __restrict__ base,
                                                int t, int k0) {
    Frag f;
    f.u[0] = *reinterpret_cast<const uint4*>(&base[t * KC + k0]);
    f.u[1] = *reinterpret_cast<const uint4*>(&base[t * KC + k0 + 8]);
    return f.bf;
}

static __device__ __forceinline__ v8f wmma_bf16(v16bf a, v16bf b, v8f c) {
    return __builtin_amdgcn_wmma_f32_16x16x32_bf16(false, a, false, b, (short)0, c, false, false);
}

// issue AOPS(=12) async copies per thread for one K-chunk; addresses are clamped
// so the per-wave outstanding-op count is EXEC-uniform (pads zeroed post-wait)
static __device__ __forceinline__ void stage_async(
    unsigned short* sAhi, unsigned short* sAlo,
    unsigned short* sBhi, unsigned short* sBlo,
    const unsigned short* bhi, const unsigned short* blo,
    const unsigned short* yhi, const unsigned short* ylo,
    int tid, int f0, int t0, int b, int kc) {
#pragma unroll
    for (int i = 0; i < 2; ++i) {                 // A: 64 rows x 8 chunks = 512
        int idx = tid + i * 256;
        int row = idx >> 3;
        int k8  = (idx & 7) * 8;
        int fg  = (row < 32) ? (f0 + row) : (NFREQ + f0 + (row - 32));
        if (fg > NROWS - 1) fg = NROWS - 1;       // clamp; zeroed later
        size_t go = (size_t)fg * FL + kc + k8;
        int lo = row * KC + k8;
        copy16(&sAhi[lo], bhi + go);
        copy16(&sAlo[lo], blo + go);
    }
#pragma unroll
    for (int i = 0; i < 4; ++i) {                 // B: 128 rows x 8 chunks = 1024
        int idx = tid + i * 256;
        int t   = idx >> 3;
        int k8  = (idx & 7) * 8;
        int s   = (t0 + t) * HOP + kc + k8 - LPAD;
        if (s < 0) s = 0;                         // clamp; zeroed later
        size_t go = (size_t)b * SAMPLES + s;
        int lo = t * KC + k8;
        copy16(&sBhi[lo], yhi + go);
        copy16(&sBlo[lo], ylo + go);
    }
}

static __device__ __forceinline__ void zero_pad(
    unsigned short* sAhi, unsigned short* sAlo,
    unsigned short* sBhi, unsigned short* sBlo,
    int tid, int f0, int t0, int kc) {
#pragma unroll
    for (int i = 0; i < 2; ++i) {
        int idx = tid + i * 256;
        int row = idx >> 3;
        int k8  = (idx & 7) * 8;
        int fg  = (row < 32) ? (f0 + row) : (NFREQ + f0 + (row - 32));
        int lo  = row * KC + k8;
        if (fg >= NROWS) { zero16(&sAhi[lo]); zero16(&sAlo[lo]); }
    }
#pragma unroll
    for (int i = 0; i < 4; ++i) {
        int idx = tid + i * 256;
        int t   = idx >> 3;
        int k8  = (idx & 7) * 8;
        int s   = (t0 + t) * HOP + kc + k8 - LPAD;
        int lo  = t * KC + k8;
        if (s < 0) { zero16(&sBhi[lo]); zero16(&sBlo[lo]); }
    }
}

// one K-chunk of bf16x3 WMMA: hi*hi + hi*lo + lo*hi; real & imag share B frags
static __device__ __forceinline__ void compute_chunk(
    const unsigned short* sAhi, const unsigned short* sAlo,
    const unsigned short* sBhi, const unsigned short* sBlo,
    int rowR, int rowI, int aK0, int bK0, int nbase, int lr,
    v8f (&accR)[2], v8f (&accI)[2]) {
#pragma unroll
    for (int ks = 0; ks < KC / 32; ++ks) {
        const int ka = aK0 + ks * 32;
        const int kb = bK0 + ks * 32;
        v16bf AhR = ldfragA(sAhi, rowR, ka);
        v16bf AlR = ldfragA(sAlo, rowR, ka);
        v16bf AhI = ldfragA(sAhi, rowI, ka);
        v16bf AlI = ldfragA(sAlo, rowI, ka);
#pragma unroll
        for (int ns = 0; ns < 2; ++ns) {
            int t = nbase + ns * 16 + lr;
            v16bf Bh = ldfragB(sBhi, t, kb);
            v16bf Bl = ldfragB(sBlo, t, kb);
            accR[ns] = wmma_bf16(AhR, Bh, accR[ns]);
            accR[ns] = wmma_bf16(AhR, Bl, accR[ns]);
            accR[ns] = wmma_bf16(AlR, Bh, accR[ns]);
            accI[ns] = wmma_bf16(AhI, Bh, accI[ns]);
            accI[ns] = wmma_bf16(AhI, Bl, accI[ns]);
            accI[ns] = wmma_bf16(AlI, Bh, accI[ns]);
        }
    }
}

// ---------------- kernel 1: one-shot fp32 -> hi/lo bf16 planes ----------------
__global__ __launch_bounds__(256)
void preconvert_kernel(const float* __restrict__ y, const float* __restrict__ basis,
                       unsigned short* __restrict__ ws) {
    unsigned short* yhi = ws;
    unsigned short* ylo = ws + YN;
    unsigned short* bhi = ws + 2 * (size_t)YN;
    unsigned short* blo = ws + 2 * (size_t)YN + BN;
    const int total4 = (YN + BN) / 4;
    for (int i = blockIdx.x * blockDim.x + threadIdx.x; i < total4;
         i += gridDim.x * blockDim.x) {
        const float* src;
        unsigned short *dhi, *dlo;
        size_t e;
        if (i < YN / 4) { e = (size_t)i * 4; src = y + e;     dhi = yhi + e; dlo = ylo + e; }
        else            { e = (size_t)(i - YN / 4) * 4; src = basis + e; dhi = bhi + e; dlo = blo + e; }
        float4 v = *reinterpret_cast<const float4*>(src);
        uint2 hp, lp;
        split4(v, hp, lp);
        *reinterpret_cast<uint2*>(dhi) = hp;
        *reinterpret_cast<uint2*>(dlo) = lp;
    }
}

// ---------------- kernel 2: bf16x3 WMMA GEMM + mag/phase epilogue ----------------
template <bool PRECONV>
__global__ __launch_bounds__(256)
void stft_wmma_kernel(const float* __restrict__ y, const float* __restrict__ basis,
                      const unsigned short* __restrict__ ws, float* __restrict__ out) {
    // double-buffered LDS: 2 x (8+8+16+16) KB = 96 KB of the 320 KB WGP pool
    __shared__ __align__(16) unsigned short sAhi[2][64 * KC];
    __shared__ __align__(16) unsigned short sAlo[2][64 * KC];
    __shared__ __align__(16) unsigned short sBhi[2][NT * KC];
    __shared__ __align__(16) unsigned short sBlo[2][NT * KC];

    const int tid   = threadIdx.x;
    const int lane  = tid & 31;
    const int wave  = tid >> 5;
    const int waveM = wave >> 2;     // 0..1
    const int waveN = wave & 3;      // 0..3

    const int b  = blockIdx.z;
    const int f0 = blockIdx.y * MT;
    const int t0 = blockIdx.x * NT;

    const int lr    = lane & 15;
    const int aK0   = (lane < 16) ? 0 : 16;
    const int bK0   = (lane < 16) ? 0 : 16;
    const int rowR  = waveM * 16 + lr;
    const int rowI  = 32 + rowR;
    const int nbase = waveN * 32;

    const unsigned short* yhi = ws;
    const unsigned short* ylo = ws + YN;
    const unsigned short* bhi = ws + 2 * (size_t)YN;
    const unsigned short* blo = ws + 2 * (size_t)YN + BN;

    v8f accR[2], accI[2];
#pragma unroll
    for (int i = 0; i < 8; ++i) {
        accR[0][i] = 0.f; accR[1][i] = 0.f;
        accI[0][i] = 0.f; accI[1][i] = 0.f;
    }

    const bool hasPadA = (f0 + MT > NFREQ);      // only last freq tile
    if (PRECONV) {
        // software pipeline: async-stage chunk i+1 while computing chunk i
        stage_async(sAhi[0], sAlo[0], sBhi[0], sBlo[0],
                    bhi, blo, yhi, ylo, tid, f0, t0, b, 0);
        for (int i = 0; i < NITER; ++i) {
            const int kc  = i * KC;
            const int cur = i & 1;
            if (i + 1 < NITER) {
                const int nxt = cur ^ 1;
                stage_async(sAhi[nxt], sAlo[nxt], sBhi[nxt], sBlo[nxt],
                            bhi, blo, yhi, ylo, tid, f0, t0, b, kc + KC);
                wait_asyncN<AOPS>();             // chunk i landed; i+1 in flight
            } else {
                wait_asyncN<0>();
            }
            if (hasPadA || (t0 == 0 && kc < LPAD))
                zero_pad(sAhi[cur], sAlo[cur], sBhi[cur], sBlo[cur],
                         tid, f0, t0, kc);
            __syncthreads();
            compute_chunk(sAhi[cur], sAlo[cur], sBhi[cur], sBlo[cur],
                          rowR, rowI, aK0, bK0, nbase, lr, accR, accI);
            __syncthreads();                     // readers done before overwrite
        }
    } else {
        // fallback: inline fp32 -> hi/lo split while staging (single buffer)
        for (int kc = 0; kc < FL; kc += KC) {
#pragma unroll
            for (int i = 0; i < 4; ++i) {        // A: 64 rows x 16 f4 = 1024
                int idx = tid + i * 256;
                int row = idx >> 4;
                int k4  = (idx & 15) * 4;
                int fg  = (row < 32) ? (f0 + row) : (NFREQ + f0 + (row - 32));
                float4 v = make_float4(0.f, 0.f, 0.f, 0.f);
                if (fg < NROWS)
                    v = *reinterpret_cast<const float4*>(&basis[(size_t)fg * FL + kc + k4]);
                uint2 hp, lp; split4(v, hp, lp);
                *reinterpret_cast<uint2*>(&sAhi[0][row * KC + k4]) = hp;
                *reinterpret_cast<uint2*>(&sAlo[0][row * KC + k4]) = lp;
            }
#pragma unroll
            for (int i = 0; i < 8; ++i) {        // B: 128 rows x 16 f4 = 2048
                int idx = tid + i * 256;
                int t   = idx >> 4;
                int k4  = (idx & 15) * 4;
                int s   = (t0 + t) * HOP + kc + k4 - LPAD;
                float4 v = make_float4(0.f, 0.f, 0.f, 0.f);
                if (s >= 0)
                    v = *reinterpret_cast<const float4*>(&y[(size_t)b * SAMPLES + s]);
                uint2 hp, lp; split4(v, hp, lp);
                *reinterpret_cast<uint2*>(&sBhi[0][t * KC + k4]) = hp;
                *reinterpret_cast<uint2*>(&sBlo[0][t * KC + k4]) = lp;
            }
            __syncthreads();
            compute_chunk(sAhi[0], sAlo[0], sBhi[0], sBlo[0],
                          rowR, rowI, aK0, bK0, nbase, lr, accR, accI);
            __syncthreads();
        }
    }

    // ---- epilogue: magnitude + phase (this VALU is the requested math) ----
#pragma unroll
    for (int ns = 0; ns < 2; ++ns) {
        int t = t0 + nbase + ns * 16 + lr;
#pragma unroll
        for (int r = 0; r < 8; ++r) {
            int f = f0 + waveM * 16 + r + ((lane >> 4) << 3);
            if (f < NFREQ) {
                float rr = accR[ns][r];
                float ii = accI[ns][r];
                size_t o = ((size_t)b * NFREQ + f) * NFRAMES + t;
                out[o]         = sqrtf(rr * rr + ii * ii);
                out[MAGSZ + o] = atan2f(ii, rr);
            }
        }
    }
}

extern "C" void kernel_launch(void* const* d_in, const int* in_sizes, int n_in,
                              void* d_out, int out_size, void* d_ws, size_t ws_size,
                              hipStream_t stream) {
    const float* y     = (const float*)d_in[0];   // (16, 262144) f32
    const float* basis = (const float*)d_in[1];   // (1026, 1, 1024) f32
    float* out = (float*)d_out;

    dim3 grid(NFRAMES / NT, (NFREQ + MT - 1) / MT, NBATCH);   // (8, 17, 16)

    if (ws_size >= (size_t)WS_ELEMS * sizeof(unsigned short)) {
        unsigned short* ws = (unsigned short*)d_ws;
        preconvert_kernel<<<1024, 256, 0, stream>>>(y, basis, ws);
        stft_wmma_kernel<true><<<grid, 256, 0, stream>>>(y, basis, ws, out);
    } else {
        stft_wmma_kernel<false><<<grid, 256, 0, stream>>>(y, basis, nullptr, out);
    }
}
